// PM25TransformerModel_12670153523464
// MI455X (gfx1250) — compile-verified, matrix operand order checked by reading, not run on hardware
//
#include <hip/hip_runtime.h>
#include <hip/hip_bf16.h>

typedef __attribute__((ext_vector_type(16))) _Float16 v16h;
typedef __attribute__((ext_vector_type(8)))  float    v8f;
typedef unsigned int  u32;
typedef unsigned long long u64;

#define Bsz   1024
#define Tlen  36
#define Snum  210
#define Spad  224          // Snum padded to multiple of 32 (zero-filled)
#define Edim  384
#define Hh    6
#define Dd    64
#define Ll    6
#define E4    (4*Edim)
#define BT    (Bsz*Tlen)

// ---------------------------------------------------------------------------
// WMMA GEMM:  C[M,N] = A[M,K] (f16 row-major) * Bt[N,K]^T (f16)
// Tile 128x64, K-step 32, 256 threads = 8 waves; wave (mw,nw) owns 32x32 of C
// -> 4 x v_wmma_f32_16x16x32_f16 per K-step per wave.
// Double-buffered staging via global_load_async_to_lds_b128 (ASYNCcnt);
// last K-iteration peeled so the hot loop prefetches unconditionally.
// Requires: M % 128 == 0, N % 64 == 0, K % 32 == 0 (buffers pre-padded).
// ---------------------------------------------------------------------------
#define TILE_K 32
#define LPITCH 40                    // halves; 80B row pitch, 16B aligned
#define BQOFF  (128 * LPITCH)        // halves offset of B region within a buffer
#define BUFH   ((128 + 64) * LPITCH) // halves per double-buffer slot
#define GEMM_LDS_BYTES (2 * BUFH * 2)

__device__ __forceinline__ void async_tile(unsigned ldsA0, unsigned ldsA1, unsigned ldsB,
                                           const _Float16* a0, const _Float16* a1,
                                           const _Float16* b)
{
    asm volatile("global_load_async_to_lds_b128 %0, %1, off" :: "v"(ldsA0), "v"(a0) : "memory");
    asm volatile("global_load_async_to_lds_b128 %0, %1, off" :: "v"(ldsA1), "v"(a1) : "memory");
    asm volatile("global_load_async_to_lds_b128 %0, %1, off" :: "v"(ldsB),  "v"(b)  : "memory");
}

template<bool HAS_BIAS, bool RELU, bool RES, bool OUT32, bool OUT16>
__global__ __launch_bounds__(256) void gemm_wmma(
    const _Float16* __restrict__ A,
    const _Float16* __restrict__ Bt,
    const float* __restrict__ bias,
    const float* __restrict__ Cin,
    float* __restrict__ Cout32,
    _Float16* __restrict__ Cout16,
    int M, int N, int K)
{
    extern __shared__ __attribute__((aligned(16))) _Float16 smem[];

    const int tid  = threadIdx.x;
    const int wave = tid >> 5;
    const int lane = tid & 31;
    const int mw   = wave >> 1;     // 0..3  (M direction, 32 rows each)
    const int nw   = wave & 1;      // 0..1  (N direction, 32 cols each)
    const int rowA0 = blockIdx.x * 128;
    const int rowB0 = blockIdx.y * 64;

    // staging: A = 128 rows x 32 halves -> 512 x 16B chunks (2/thread)
    //          B =  64 rows x 32 halves -> 256 x 16B chunks (1/thread)
    const int arow = tid >> 2;                 // 0..63
    const int acol = (tid & 3) * 8;            // 0/8/16/24
    const unsigned ldsA0 = (unsigned)(( arow       * LPITCH + acol) * 2);
    const unsigned ldsA1 = (unsigned)(((arow + 64) * LPITCH + acol) * 2);
    const unsigned ldsB  = (unsigned)((BQOFF + arow * LPITCH + acol) * 2);
    const unsigned bufB  = BUFH * 2;           // bytes per LDS buffer slot

    const _Float16* pA0 = A  + (size_t)(rowA0 + arow)      * K + acol;
    const _Float16* pA1 = A  + (size_t)(rowA0 + arow + 64) * K + acol;
    const _Float16* pB  = Bt + (size_t)(rowB0 + arow)      * K + acol;

    v8f acc[2][2] = {{{}, {}}, {{}, {}}};

    const int l15  = lane & 15;
    const int koff = (lane >> 4) ? 8 : 0;

    // fragment load + 4 WMMAs from buffer slot `sel`
    auto compute = [&](int sel) {
        const _Float16* As = smem + sel * BUFH;
        const _Float16* Bs = As + BQOFF;
        // fragments per ISA 16-bit 16x32 layout:
        // lanes 0-15: K {0..7,16..23}; lanes 16-31: K {8..15,24..31}
        union Frag { v16h v; uint4 u[2]; } af[2], bf[2];
#pragma unroll
        for (int f = 0; f < 2; f++) {
            const _Float16* ap = &As[(mw * 32 + f * 16 + l15) * LPITCH + koff];
            af[f].u[0] = *(const uint4*)ap;
            af[f].u[1] = *(const uint4*)(ap + 16);
            const _Float16* bp = &Bs[(nw * 32 + f * 16 + l15) * LPITCH + koff];
            bf[f].u[0] = *(const uint4*)bp;
            bf[f].u[1] = *(const uint4*)(bp + 16);
        }
#pragma unroll
        for (int fm = 0; fm < 2; fm++)
#pragma unroll
            for (int fn = 0; fn < 2; fn++)
                acc[fm][fn] = __builtin_amdgcn_wmma_f32_16x16x32_f16(
                    false, af[fm].v, false, bf[fn].v,
                    (short)0, acc[fm][fn], false, false);
    };

    // prime buffer 0 with K-tile 0
    async_tile(ldsA0, ldsA1, ldsB, pA0, pA1, pB);

    const int nk = K >> 5;
    int sel = 0;
    for (int i = 0; i < nk - 1; i++) {
        // unconditionally prefetch next tile into the other buffer
        const unsigned so = (sel ^ 1) * bufB;
        async_tile(ldsA0 + so, ldsA1 + so, ldsB + so,
                   pA0 + TILE_K, pA1 + TILE_K, pB + TILE_K);
        pA0 += TILE_K; pA1 += TILE_K; pB += TILE_K;
        asm volatile("s_wait_asynccnt 3" ::: "memory");  // tile i landed
        __syncthreads();
        compute(sel);
        __syncthreads();
        sel ^= 1;
    }
    // peeled final tile: drain and compute, no further prefetch
    asm volatile("s_wait_asynccnt 0" ::: "memory");
    __syncthreads();
    compute(sel);

    // epilogue (fully branchless per template): lane l, vgpr r ->
    // M = (l>>4)*8 + r, N = l&15
    const int lh8 = (lane >> 4) * 8;
#pragma unroll
    for (int fm = 0; fm < 2; fm++) {
        const int grow0 = rowA0 + mw * 32 + fm * 16 + lh8;
#pragma unroll
        for (int fn = 0; fn < 2; fn++) {
            const int col = rowB0 + nw * 32 + fn * 16 + l15;
            const float bv = HAS_BIAS ? bias[col] : 0.f;
#pragma unroll
            for (int r = 0; r < 8; r++) {
                const size_t o = (size_t)(grow0 + r) * N + col;
                float v = acc[fm][fn][r] + bv;
                if (RELU)  v = fmaxf(v, 0.f);
                if (RES)   v += Cin[o];
                if (OUT32) Cout32[o] = v;
                if (OUT16) Cout16[o] = (_Float16)v;
            }
        }
    }
}

// ---------------------------------------------------------------------------
// Weight convert+transpose: in[K,N] f32 -> out[N,K] f16   (K % 32 == 0 cases)
// ---------------------------------------------------------------------------
__global__ void convert_wt(const float* __restrict__ in, _Float16* __restrict__ out,
                           int K, int N)
{
    int i = blockIdx.x * 256 + threadIdx.x;
    if (i >= K * N) return;
    int n = i / K, k = i - n * K;
    out[i] = (_Float16)in[(size_t)k * N + n];
}

// Wp (Snum x E) f32 -> out (E x Spad) f16, zero-padded K
__global__ void wp_convert(const float* __restrict__ Wp, _Float16* __restrict__ out)
{
    int i = blockIdx.x * 256 + threadIdx.x;
    if (i >= Edim * Spad) return;
    int n = i / Spad, k = i - n * Spad;
    out[i] = (k < Snum) ? (_Float16)Wp[(size_t)k * Edim + n] : (_Float16)0.f;
}

// norm pm25 -> f16, padded to Spad columns: pm[bt*Spad+s]
__global__ void pm_convert(const int* __restrict__ idx, _Float16* __restrict__ pm)
{
    int i = blockIdx.x * 256 + threadIdx.x;
    if (i >= BT * Spad) return;
    int bt = i / Spad, s = i - bt * Spad;
    pm[i] = (s < Snum)
          ? (_Float16)((float)idx[((size_t)bt * Snum + s) * 2] * (1.0f / 500.0f))
          : (_Float16)0.f;
}

// x += fire_emb[fire] + station_emb[stn] + sinusoidal_pe(t, n)
__global__ void embed_add(float* __restrict__ x, const int* __restrict__ idx,
                          const int* __restrict__ stn_ix_p,
                          const float* __restrict__ fire_emb,
                          const float* __restrict__ station_emb)
{
    int i = blockIdx.x * 256 + threadIdx.x;    // over BT*E
    int bt = i / Edim, n = i - bt * Edim;
    int t  = bt % Tlen;
    int stn  = *stn_ix_p;
    int fire = idx[((size_t)bt * Snum + stn) * 2 + 1];
    float freq = __expf(-(float)(n & ~1) * (9.210340371976184f / (float)Edim));
    float ang  = (float)t * freq;
    float pe   = (n & 1) ? __cosf(ang) : __sinf(ang);
    x[i] += fire_emb[fire * Edim + n] + station_emb[stn * Edim + n] + pe;
}

// ---------------------------------------------------------------------------
// LayerNorm: one wave per row of 384; 8 rows/block. f32 in, f16 out
// ---------------------------------------------------------------------------
__global__ __launch_bounds__(256) void ln_kernel(const float* __restrict__ x,
    const float* __restrict__ g, const float* __restrict__ b,
    _Float16* __restrict__ out, int nrows)
{
    const int wave = threadIdx.x >> 5, lane = threadIdx.x & 31;
    const int row  = blockIdx.x * 8 + wave;
    if (row >= nrows) return;
    const float* xr = x + (size_t)row * Edim;
    float vals[12], s = 0.f, sq = 0.f;
#pragma unroll
    for (int i = 0; i < 12; i++) {
        float v = xr[lane + i * 32];
        vals[i] = v; s += v; sq += v * v;
    }
#pragma unroll
    for (int m = 16; m >= 1; m >>= 1) {
        s  += __shfl_xor(s,  m, 32);
        sq += __shfl_xor(sq, m, 32);
    }
    float mean = s * (1.0f / Edim);
    float var  = sq * (1.0f / Edim) - mean * mean;
    float rstd = rsqrtf(var + 1e-5f);
    _Float16* o = out + (size_t)row * Edim;
#pragma unroll
    for (int i = 0; i < 12; i++) {
        int e = lane + i * 32;
        o[e] = (_Float16)((vals[i] - mean) * rstd * g[e] + b[e]);
    }
}

// ---------------------------------------------------------------------------
// Attention: one block per (b,h); T=36, D=64, causal softmax.
// ---------------------------------------------------------------------------
__global__ __launch_bounds__(64) void attn_kernel(
    const _Float16* __restrict__ q, const _Float16* __restrict__ k,
    const _Float16* __restrict__ v, _Float16* __restrict__ o)
{
    const int bh = blockIdx.x;
    const int b  = bh / Hh, h = bh - b * Hh;
    __shared__ _Float16 qs[Tlen * Dd], ks[Tlen * Dd], vs[Tlen * Dd];
    __shared__ float sc[Tlen][40];
    const size_t base = ((size_t)b * Tlen) * Edim + h * Dd;

    for (int i = threadIdx.x; i < (Tlen * Dd) / 4; i += 64) {
        int rr = i >> 4, cc = (i & 15) << 2;
        size_t go = base + (size_t)rr * Edim + cc;
        *(u64*)&qs[rr * Dd + cc] = *(const u64*)&q[go];
        *(u64*)&ks[rr * Dd + cc] = *(const u64*)&k[go];
        *(u64*)&vs[rr * Dd + cc] = *(const u64*)&v[go];
    }
    __syncthreads();

    const int t = threadIdx.x;
    if (t < Tlen) {
        float mx = -1e30f;
        for (int j = 0; j <= t; j++) {
            float d = 0.f;
#pragma unroll
            for (int dd = 0; dd < Dd; dd++)
                d += (float)qs[t * Dd + dd] * (float)ks[j * Dd + dd];
            d *= 0.125f;            // D^-0.5
            sc[t][j] = d;
            mx = fmaxf(mx, d);
        }
        float sum = 0.f;
        for (int j = 0; j <= t; j++) {
            float e = __expf(sc[t][j] - mx);
            sc[t][j] = e; sum += e;
        }
        const float inv = 1.0f / sum;
        float ob[Dd];
#pragma unroll
        for (int dd = 0; dd < Dd; dd++) ob[dd] = 0.f;
        for (int j = 0; j <= t; j++) {
            float w = sc[t][j] * inv;
#pragma unroll
            for (int dd = 0; dd < Dd; dd++) ob[dd] += w * (float)vs[j * Dd + dd];
        }
        _Float16* op = o + base + (size_t)t * Edim;
#pragma unroll
        for (int dd = 0; dd < Dd; dd++) op[dd] = (_Float16)ob[dd];
    }
}

// ---------------------------------------------------------------------------
// Final LN + head (E->1) + sigmoid; writes logits (f32) and denorm to d_out
// ---------------------------------------------------------------------------
__global__ __launch_bounds__(256) void head_kernel(const float* __restrict__ x,
    const float* __restrict__ g, const float* __restrict__ b,
    const float* __restrict__ Wh, const float* __restrict__ bh,
    float* __restrict__ logits, float* __restrict__ denorm, int nrows)
{
    const int wave = threadIdx.x >> 5, lane = threadIdx.x & 31;
    const int row  = blockIdx.x * 8 + wave;
    if (row >= nrows) return;
    const float* xr = x + (size_t)row * Edim;
    float vals[12], s = 0.f, sq = 0.f;
#pragma unroll
    for (int i = 0; i < 12; i++) {
        float v = xr[lane + i * 32];
        vals[i] = v; s += v; sq += v * v;
    }
#pragma unroll
    for (int m = 16; m >= 1; m >>= 1) {
        s  += __shfl_xor(s,  m, 32);
        sq += __shfl_xor(sq, m, 32);
    }
    float mean = s * (1.0f / Edim);
    float rstd = rsqrtf(sq * (1.0f / Edim) - mean * mean + 1e-5f);
    float dot = 0.f;
#pragma unroll
    for (int i = 0; i < 12; i++) {
        int e = lane + i * 32;
        dot += ((vals[i] - mean) * rstd * g[e] + b[e]) * Wh[e];
    }
#pragma unroll
    for (int m = 16; m >= 1; m >>= 1) dot += __shfl_xor(dot, m, 32);
    if (lane == 0) {
        float p = 1.0f / (1.0f + __expf(-(dot + bh[0])));
        logits[row] = p;
        denorm[row] = p * 500.0f;
    }
}

__global__ void init_sums(float* sums) { if (threadIdx.x < 4) sums[threadIdx.x] = 0.f; }

__global__ __launch_bounds__(256) void reduce_kernel(const float* __restrict__ logits,
    const int* __restrict__ targets, float* __restrict__ sums, int n)
{
    int i = blockIdx.x * 256 + threadIdx.x;
    float a1 = 0.f, a2 = 0.f, a3 = 0.f, a4 = 0.f;
    if (i < n) {
        float l  = logits[i];
        float y  = (float)targets[2 * i];
        float nt = y * (1.0f / 500.0f);
        float d1 = l - nt;         a1 = d1 * d1;
        a2 = y; a3 = y * y;
        float d4 = y - l * 500.0f; a4 = d4 * d4;
    }
#pragma unroll
    for (int m = 16; m >= 1; m >>= 1) {
        a1 += __shfl_xor(a1, m, 32); a2 += __shfl_xor(a2, m, 32);
        a3 += __shfl_xor(a3, m, 32); a4 += __shfl_xor(a4, m, 32);
    }
    if ((threadIdx.x & 31) == 0) {
        atomicAdd(&sums[0], a1); atomicAdd(&sums[1], a2);
        atomicAdd(&sums[2], a3); atomicAdd(&sums[3], a4);
    }
}

__global__ void finalize_kernel(const float* __restrict__ sums, float* __restrict__ out2, int n)
{
    float s1 = sums[0], s2 = sums[1], s3 = sums[2], s4 = sums[3];
    out2[0] = s1 / (float)n;                       // mse
    float ss = s3 - s2 * s2 / (float)n;            // sum (y - mean y)^2
    out2[1] = 1.0f - s4 / ss;                      // r2
}

// ---------------------------------------------------------------------------
template<bool HB, bool RL, bool RS, bool O32, bool O16>
static inline void launch_gemm(const _Float16* A, const _Float16* Bt, const float* bias,
                               const float* Cin, float* C32, _Float16* C16,
                               int M, int N, int K, hipStream_t s)
{
    gemm_wmma<HB, RL, RS, O32, O16>
        <<<dim3(M / 128, N / 64), dim3(256), GEMM_LDS_BYTES, s>>>(
            A, Bt, bias, Cin, C32, C16, M, N, K);
}

extern "C" void kernel_launch(void* const* d_in, const int* in_sizes, int n_in,
                              void* d_out, int out_size, void* d_ws, size_t ws_size,
                              hipStream_t stream)
{
    const int*   idx     = (const int*)  d_in[0];
    const int*   targets = (const int*)  d_in[1];
    const int*   stn_ix  = (const int*)  d_in[3];
    const float* Wp      = (const float*)d_in[4];
    const float* bp      = (const float*)d_in[5];
    const float* st_emb  = (const float*)d_in[6];
    const float* fi_emb  = (const float*)d_in[7];
    const float* ln1g    = (const float*)d_in[8];
    const float* ln1b    = (const float*)d_in[9];
    const float* Wq      = (const float*)d_in[10];
    const float* Wk      = (const float*)d_in[11];
    const float* Wv      = (const float*)d_in[12];
    const float* Wo      = (const float*)d_in[13];
    const float* bo      = (const float*)d_in[14];
    const float* ln2g    = (const float*)d_in[15];
    const float* ln2b    = (const float*)d_in[16];
    const float* W1      = (const float*)d_in[17];
    const float* b1      = (const float*)d_in[18];
    const float* W2      = (const float*)d_in[19];
    const float* b2      = (const float*)d_in[20];
    const float* lnfg    = (const float*)d_in[21];
    const float* lnfb    = (const float*)d_in[22];
    const float* Wh      = (const float*)d_in[23];
    const float* bh      = (const float*)d_in[24];

    char* ws = (char*)d_ws;
    size_t off = 0;
    auto carve = [&](size_t bytes) -> char* {
        char* p = ws + off;
        off += (bytes + 255) & ~(size_t)255;
        return p;
    };
    float*     x     = (float*)    carve((size_t)BT * Edim * 4);
    _Float16*  h16   = (_Float16*) carve((size_t)BT * Edim * 2);
    _Float16*  q16   = (_Float16*) carve((size_t)BT * Edim * 2);
    _Float16*  k16   = (_Float16*) carve((size_t)BT * Edim * 2);
    _Float16*  v16   = (_Float16*) carve((size_t)BT * Edim * 2);
    _Float16*  o16   = (_Float16*) carve((size_t)BT * Edim * 2);
    _Float16*  mid16 = (_Float16*) carve((size_t)BT * E4 * 2);
    _Float16*  pm16  = (_Float16*) carve((size_t)BT * Spad * 2);
    _Float16*  wp_t  = (_Float16*) carve((size_t)Edim * Spad * 2);
    _Float16*  wq_t  = (_Float16*) carve((size_t)Ll * Edim * Edim * 2);
    _Float16*  wk_t  = (_Float16*) carve((size_t)Ll * Edim * Edim * 2);
    _Float16*  wv_t  = (_Float16*) carve((size_t)Ll * Edim * Edim * 2);
    _Float16*  wo_t  = (_Float16*) carve((size_t)Ll * Edim * Edim * 2);
    _Float16*  w1_t  = (_Float16*) carve((size_t)Ll * Edim * E4 * 2);
    _Float16*  w2_t  = (_Float16*) carve((size_t)Ll * Edim * E4 * 2);
    float*     logits= (float*)    carve((size_t)BT * 4);
    float*     sums  = (float*)    carve(32);

    auto cvt = [&](const float* in, _Float16* out, int K, int N) {
        int tot = K * N;
        convert_wt<<<dim3((tot + 255) / 256), dim3(256), 0, stream>>>(in, out, K, N);
    };

    // weight conversion (transposed f16 copies)
    wp_convert<<<dim3((Edim * Spad + 255) / 256), dim3(256), 0, stream>>>(Wp, wp_t);
    for (int l = 0; l < Ll; l++) {
        cvt(Wq + (size_t)l * Edim * Edim, wq_t + (size_t)l * Edim * Edim, Edim, Edim);
        cvt(Wk + (size_t)l * Edim * Edim, wk_t + (size_t)l * Edim * Edim, Edim, Edim);
        cvt(Wv + (size_t)l * Edim * Edim, wv_t + (size_t)l * Edim * Edim, Edim, Edim);
        cvt(Wo + (size_t)l * Edim * Edim, wo_t + (size_t)l * Edim * Edim, Edim, Edim);
        cvt(W1 + (size_t)l * Edim * E4,   w1_t + (size_t)l * Edim * E4,   Edim, E4);
        cvt(W2 + (size_t)l * E4 * Edim,   w2_t + (size_t)l * E4 * Edim,   E4, Edim);
    }

    // input projection (K padded to 224 with zeros)
    pm_convert<<<dim3((BT * Spad + 255) / 256), dim3(256), 0, stream>>>(idx, pm16);
    launch_gemm<true, false, false, true, false>(
        pm16, wp_t, bp, nullptr, x, nullptr, BT, Edim, Spad, stream);
    embed_add<<<dim3((BT * Edim) / 256), dim3(256), 0, stream>>>(x, idx, stn_ix, fi_emb, st_emb);

    // transformer layers
    for (int l = 0; l < Ll; l++) {
        const size_t wo_off = (size_t)l * Edim * Edim;
        const size_t m_off  = (size_t)l * Edim * E4;
        ln_kernel<<<dim3(BT / 8), dim3(256), 0, stream>>>(x, ln1g + l * Edim, ln1b + l * Edim, h16, BT);
        launch_gemm<false, false, false, false, true>(
            h16, wq_t + wo_off, nullptr, nullptr, nullptr, q16, BT, Edim, Edim, stream);
        launch_gemm<false, false, false, false, true>(
            h16, wk_t + wo_off, nullptr, nullptr, nullptr, k16, BT, Edim, Edim, stream);
        launch_gemm<false, false, false, false, true>(
            h16, wv_t + wo_off, nullptr, nullptr, nullptr, v16, BT, Edim, Edim, stream);
        attn_kernel<<<dim3(Bsz * Hh), dim3(64), 0, stream>>>(q16, k16, v16, o16);
        launch_gemm<true, false, true, true, false>(
            o16, wo_t + wo_off, bo + l * Edim, x, x, nullptr, BT, Edim, Edim, stream);
        ln_kernel<<<dim3(BT / 8), dim3(256), 0, stream>>>(x, ln2g + l * Edim, ln2b + l * Edim, h16, BT);
        launch_gemm<true, true, false, false, true>(
            h16, w1_t + m_off, b1 + l * E4, nullptr, nullptr, mid16, BT, E4, Edim, stream);
        launch_gemm<true, false, true, true, false>(
            mid16, w2_t + m_off, b2 + l * Edim, x, x, nullptr, BT, Edim, E4, stream);
    }

    // final LN + head + losses
    head_kernel<<<dim3(BT / 8), dim3(256), 0, stream>>>(
        x, lnfg, lnfb, Wh, bh, logits, (float*)d_out, BT);
    init_sums<<<dim3(1), dim3(32), 0, stream>>>(sums);
    reduce_kernel<<<dim3(BT / 256), dim3(256), 0, stream>>>(logits, targets, sums, BT);
    finalize_kernel<<<dim3(1), dim3(1), 0, stream>>>(sums, (float*)d_out + BT, BT);
}
